// Attention_block_12704513261692
// MI455X (gfx1250) — matrix operand order repaired and checked by reading, not compiled, not run on hardware
//
#include <hip/hip_runtime.h>

typedef __attribute__((ext_vector_type(16))) _Float16 v16h;
typedef __attribute__((ext_vector_type(8)))  _Float16 v8h;
typedef __attribute__((ext_vector_type(8)))  float    v8f;
typedef __attribute__((ext_vector_type(4)))  unsigned int u32x4;
typedef __attribute__((ext_vector_type(8)))  unsigned int u32x8;

#define BATCH 4
#define CCH   256
#define NTOK  4096
#define C3    768
#define BN_EPS 1e-5f
#define ATT_SCALE 0.0625f   // 1/sqrt(256)

// ---------------- WMMA helpers (gfx1250 wave32) ----------------

__device__ __forceinline__ v8f wmma_f16(v16h a, v16h b, v8f c) {
  // D = A(16x32 f16) * B(32x16 f16) + C(16x16 f32)
  return __builtin_amdgcn_wmma_f32_16x16x32_f16(
      /*neg_a=*/false, a, /*neg_b=*/false, b,
      /*c_mod=*/(short)0, c, /*reuse_a=*/false, /*reuse_b=*/false);
}

// A fragment: per lane, halves 0..7 at p[0..7], halves 8..15 at p[8..15]
// (caller already applied the half-wave +16 offset for lanes 16..31)
__device__ __forceinline__ v16h make_a(const _Float16* p) {
  v8h lo = *(const v8h*)(p);
  v8h hi = *(const v8h*)(p + 8);
  v16h a;
#pragma unroll
  for (int i = 0; i < 8; ++i) { a[i] = lo[i]; a[i + 8] = hi[i]; }
  return a;
}

// ---------------- TDM: async 2D tile DMA global -> LDS (cdna5 §8) ---------
// Builds a 2-group D# (2D tensor, data_size = 2 bytes) in SGPRs and issues
// tensor_load_to_lds. Tracked on TENSORcnt. All args must be wave-uniform.

__device__ __forceinline__ void tdm_load_2d(unsigned lds_off,
                                            const _Float16* gaddr,
                                            unsigned tensor_d0, unsigned tensor_d1,
                                            unsigned tile_d0, unsigned tile_d1,
                                            unsigned d0_stride) {
  unsigned long long ga = (unsigned long long)gaddr;
  u32x4 g0;
  g0[0] = 1u;                                   // count=1, user descriptor
  g0[1] = lds_off;                              // LDS byte address
  g0[2] = (unsigned)(ga & 0xffffffffu);         // global addr [31:0]
  g0[3] = (unsigned)((ga >> 32) & 0x01ffffffu)  // global addr [56:32]
          | (2u << 30);                         // type = 2 ("image")
  u32x8 g1;
  g1[0] = 1u << 16;                             // data_size = 1 -> 2 bytes
  g1[1] = tensor_d0 << 16;                      // tensor_dim0 [15:0]
  g1[2] = (tensor_d0 >> 16) | (tensor_d1 << 16);
  g1[3] = (tensor_d1 >> 16) | (tile_d0 << 16);  // tile_dim0
  g1[4] = tile_d1 & 0xffffu;                    // tile_dim1 (tile_dim2 = 0)
  g1[5] = d0_stride;                            // tensor_dim0_stride [31:0]
  g1[6] = 0;                                    // stride hi / dim1_stride lo
  g1[7] = 0;
  asm volatile("tensor_load_to_lds %0, %1" :: "s"(g0), "s"(g1) : "memory");
}

// ---------------- Kernel 1: BN statistics -> per-channel scale/shift ------

__global__ void bn_stats_kernel(const float* __restrict__ x,
                                const float* __restrict__ gamma,
                                const float* __restrict__ beta,
                                float* __restrict__ scale,
                                float* __restrict__ shift) {
  __shared__ float ssum[256], ssq[256];
  const int ch = blockIdx.x, tid = threadIdx.x;
  float s = 0.f, s2 = 0.f;
  for (int b = 0; b < BATCH; ++b) {
    const float* p = x + (size_t)(b * CCH + ch) * NTOK;
    for (int i = tid; i < NTOK; i += 256) {
      float v = p[i];
      s += v; s2 += v * v;
    }
  }
  ssum[tid] = s; ssq[tid] = s2;
  __syncthreads();
  for (int off = 128; off > 0; off >>= 1) {
    if (tid < off) { ssum[tid] += ssum[tid + off]; ssq[tid] += ssq[tid + off]; }
    __syncthreads();
  }
  if (tid == 0) {
    const float n = (float)(BATCH * NTOK);
    float mean = ssum[0] / n;
    float var  = ssq[0] / n - mean * mean;
    float inv  = rsqrtf(var + BN_EPS);
    float sc   = gamma[ch] * inv;
    scale[ch]  = sc;
    shift[ch]  = beta[ch] - mean * sc;
  }
}

// ---------------- Kernel 2: normalize + transpose [b,c,n] -> f16 [b,n,c] --

__global__ void bn_transpose_kernel(const float* __restrict__ x,
                                    const float* __restrict__ scale,
                                    const float* __restrict__ shift,
                                    _Float16* __restrict__ xn) {
  __shared__ float tile[32][33];
  const int t0 = blockIdx.x * 32, c0 = blockIdx.y * 32, b = blockIdx.z;
  const int tx = threadIdx.x, ty = threadIdx.y;
#pragma unroll
  for (int i = 0; i < 4; ++i) {
    int cl = ty + 8 * i;
    int ch = c0 + cl;
    float v = x[(size_t)(b * CCH + ch) * NTOK + t0 + tx];
    tile[cl][tx] = v * scale[ch] + shift[ch];
  }
  __syncthreads();
#pragma unroll
  for (int i = 0; i < 4; ++i) {
    int tl = ty + 8 * i;
    xn[((size_t)(b * NTOK + t0 + tl)) * CCH + c0 + tx] = (_Float16)tile[tx][tl];
  }
}

// ---------------- Kernel 2b: weights fp32 -> f16 --------------------------

__global__ void cvt_w_kernel(const float* __restrict__ wqkv,
                             const float* __restrict__ wout,
                             _Float16* __restrict__ wqkv_h,
                             _Float16* __restrict__ wout_h) {
  int i = blockIdx.x * 256 + threadIdx.x;
  if (i < C3 * CCH) wqkv_h[i] = (_Float16)wqkv[i];
  if (i < CCH * CCH) wout_h[i] = (_Float16)wout[i];
}

// ---------------- Kernel 3: QKV projection GEMM (WMMA) --------------------

__global__ void qkv_gemm_kernel(const _Float16* __restrict__ xn,
                                const _Float16* __restrict__ wqkv,
                                const float* __restrict__ bqkv,
                                _Float16* __restrict__ qh,
                                _Float16* __restrict__ kh,
                                _Float16* __restrict__ vt) {
  const int wid  = (blockIdx.x * blockDim.x + threadIdx.x) >> 5;
  const int lane = threadIdx.x & 31;
  const int tt   = wid & 255;          // token tile
  const int rem  = wid >> 8;
  const int ot   = rem % 48;           // out-channel tile (of 768)
  const int b    = rem / 48;
  const int half = (lane >= 16);
  const int lm   = lane & 15;
  const int tok  = tt * 16 + lm;
  const int o    = ot * 16 + lm;

  const _Float16* arow = xn + ((size_t)(b * NTOK + tok)) * CCH + half * 16;
  const _Float16* brow = wqkv + (size_t)o * CCH + half * 16;

  v8f acc = {};
#pragma unroll
  for (int cc = 0; cc < 8; ++cc)
    acc = wmma_f16(make_a(arow + cc * 32), *(const v16h*)(brow + cc * 32), acc);

  const float bias = bqkv[o];
#pragma unroll
  for (int r = 0; r < 8; ++r) acc[r] += bias;

  const int trow = tt * 16 + half * 8;  // C-layout: row = r + 8*half
  if (ot < 16) {          // Q, fold in attention scale
#pragma unroll
    for (int r = 0; r < 8; ++r)
      qh[((size_t)(b * NTOK + trow + r)) * CCH + o] = (_Float16)(acc[r] * ATT_SCALE);
  } else if (ot < 32) {   // K, token-major
    const int ch = o - CCH;
#pragma unroll
    for (int r = 0; r < 8; ++r)
      kh[((size_t)(b * NTOK + trow + r)) * CCH + ch] = (_Float16)acc[r];
  } else {                // V, channel-major (transposed)
    const int ch = o - 2 * CCH;
#pragma unroll
    for (int r = 0; r < 8; ++r)
      vt[((size_t)(b * CCH + ch)) * NTOK + trow + r] = (_Float16)acc[r];
  }
}

// ---------------- Kernel 4: flash attention ------------------------------
// One wave per 16-query tile; 8 waves/block share one batch. K/V tiles are
// DMA'd into LDS by the Tensor Data Mover (double-buffered, issued by wave 0,
// synchronized with s_wait_tensorcnt + workgroup barriers).
//
// Dynamic LDS layout (halves):
//   [    0,  8192) K buf0  (32 tok x 256 ch, row-major)
//   [ 8192, 16384) K buf1
//   [16384, 24576) V buf0  (256 ch x 32 key, row-major)
//   [24576, 32768) V buf1
//   [32768, 36864) P staging, 512 halves per wave

__global__ void __launch_bounds__(256, 1)
flash_attn_kernel(const _Float16* __restrict__ qh,
                  const _Float16* __restrict__ kh,
                  const _Float16* __restrict__ vt,
                  _Float16* __restrict__ oh) {
  extern __shared__ _Float16 smem[];

  const int wslot = threadIdx.x >> 5;
  const int lane  = threadIdx.x & 31;
  const int qt    = (blockIdx.x * 8 + wslot) & 255;
  const int b     = blockIdx.x >> 5;          // 32 blocks per batch (uniform)
  const int half  = (lane >= 16);
  const int lm    = lane & 15;
  _Float16* pbuf  = smem + 32768 + wslot * 512;

  const unsigned lds0 = __builtin_amdgcn_groupstaticsize();  // dynamic LDS base

  const _Float16* kbase = kh + ((size_t)b * NTOK) * CCH;
  const _Float16* vbase = vt + ((size_t)b * CCH) * NTOK;

  // Q tile resident in registers: 8 A-fragments over K=256
  const _Float16* qrow = qh + ((size_t)(b * NTOK + qt * 16 + lm)) * CCH + half * 16;
  v16h qa[8];
#pragma unroll
  for (int cc = 0; cc < 8; ++cc) qa[cc] = make_a(qrow + cc * 32);

  v8f zero = {};
  v8f acc[16];
#pragma unroll
  for (int g = 0; g < 16; ++g) acc[g] = zero;
  float mrow[8], lrow[8];
#pragma unroll
  for (int r = 0; r < 8; ++r) { mrow[r] = -3.0e38f; lrow[r] = 0.f; }

  // prologue: DMA step-0 K/V tiles into buffer 0
  if (wslot == 0) {
    tdm_load_2d(lds0,         kbase, CCH, NTOK, CCH, 32, CCH);         // K tile
    tdm_load_2d(lds0 + 32768, vbase, NTOK, CCH, 32, CCH, NTOK);        // V tile
  }

  for (int kt = 0; kt < NTOK / 32; ++kt) {
    const int cur = kt & 1;
    const _Float16* kcur = smem + (cur ? 8192 : 0);
    const _Float16* vcur = smem + 16384 + (cur ? 8192 : 0);

    __syncthreads();   // all waves done with the buffer about to be refilled
    if (wslot == 0) {
      if (kt + 1 < NTOK / 32) {
        const unsigned nb = (cur ^ 1) ? 16384u : 0u;
        tdm_load_2d(lds0 + nb,          kbase + (size_t)(kt + 1) * 32 * CCH,
                    CCH, NTOK, CCH, 32, CCH);
        tdm_load_2d(lds0 + 32768 + nb,  vbase + (kt + 1) * 32,
                    NTOK, CCH, 32, CCH, NTOK);
        __builtin_amdgcn_s_wait_tensorcnt(2);  // step-kt tiles complete
      } else {
        __builtin_amdgcn_s_wait_tensorcnt(0);
      }
    }
    __syncthreads();   // step-kt tiles visible to all waves

    // ---- S = Q * K^T for 32 keys (two 16x16 f32 tiles), K from LDS ----
    v8f s0 = zero, s1 = zero;
    const _Float16* k0 = kcur + lm * CCH + half * 16;
    const _Float16* k1 = k0 + 16 * CCH;
#pragma unroll
    for (int cc = 0; cc < 8; ++cc) {
      s0 = wmma_f16(qa[cc], *(const v16h*)(k0 + cc * 32), s0);
      s1 = wmma_f16(qa[cc], *(const v16h*)(k1 + cc * 32), s1);
    }

    // ---- online softmax (lanes 0-15 / 16-31 hold rows 0-7 / 8-15) ----
    float sf[8];
#pragma unroll
    for (int r = 0; r < 8; ++r) {
      float a0 = s0[r], a1 = s1[r];
      float mx = fmaxf(a0, a1);
#pragma unroll
      for (int off = 1; off < 16; off <<= 1)
        mx = fmaxf(mx, __shfl_xor(mx, off, 32));
      float mnew = fmaxf(mrow[r], mx);
      float p0 = __expf(a0 - mnew);
      float p1 = __expf(a1 - mnew);
      float rs = p0 + p1;
#pragma unroll
      for (int off = 1; off < 16; off <<= 1)
        rs += __shfl_xor(rs, off, 32);
      float f = __expf(mrow[r] - mnew);
      lrow[r] = lrow[r] * f + rs;
      mrow[r] = mnew;
      sf[r]   = f;
      // stage P row-major 16x32 into this wave's LDS slot
      int row = r + half * 8;
      pbuf[row * 32 + lm]      = (_Float16)p0;
      pbuf[row * 32 + 16 + lm] = (_Float16)p1;
    }

    // rescale O accumulators by exp(m_old - m_new)
#pragma unroll
    for (int g = 0; g < 16; ++g)
#pragma unroll
      for (int r = 0; r < 8; ++r) acc[g][r] *= sf[r];

    // same-wave LDS RAW: wait for DS stores before A-fragment reload
    asm volatile("s_wait_dscnt 0" ::: "memory");

    v16h pa;
    {
      const _Float16* pp = pbuf + lm * 32 + half * 16;
      v8h lo = *(const v8h*)(pp);
      v8h hi = *(const v8h*)(pp + 8);
#pragma unroll
      for (int i = 0; i < 8; ++i) { pa[i] = lo[i]; pa[i + 8] = hi[i]; }
    }

    // ---- O += P * V over all 256 channels (16 WMMAs), V from LDS ----
    const _Float16* vp = vcur + lm * 32 + half * 16;
#pragma unroll
    for (int g = 0; g < 16; ++g)
      acc[g] = wmma_f16(pa, *(const v16h*)(vp + g * 16 * 32), acc[g]);
  }

  // ---- epilogue: O /= l, store f16 token-major ----
  float inv[8];
#pragma unroll
  for (int r = 0; r < 8; ++r) inv[r] = 1.0f / lrow[r];
#pragma unroll
  for (int g = 0; g < 16; ++g)
#pragma unroll
    for (int r = 0; r < 8; ++r) {
      int t = qt * 16 + r + half * 8;
      oh[((size_t)(b * NTOK + t)) * CCH + g * 16 + lm] = (_Float16)(acc[g][r] * inv[r]);
    }
}

// ---------------- Kernel 5: output projection + bias + residual -----------

__global__ void out_proj_kernel(const _Float16* __restrict__ oh,
                                const _Float16* __restrict__ wout,
                                const float* __restrict__ bout,
                                const float* __restrict__ x,
                                float* __restrict__ out) {
  const int wid  = (blockIdx.x * blockDim.x + threadIdx.x) >> 5;
  const int lane = threadIdx.x & 31;
  const int og   = wid & 15;
  const int rem  = wid >> 4;
  const int tt   = rem & 255;
  const int b    = rem >> 8;
  const int half = (lane >= 16);
  const int lm   = lane & 15;
  const int tok  = tt * 16 + lm;
  const int o    = og * 16 + lm;

  const _Float16* arow = oh + ((size_t)(b * NTOK + tok)) * CCH + half * 16;
  const _Float16* brow = wout + (size_t)o * CCH + half * 16;

  v8f acc = {};
#pragma unroll
  for (int cc = 0; cc < 8; ++cc)
    acc = wmma_f16(make_a(arow + cc * 32), *(const v16h*)(brow + cc * 32), acc);

  const float bias = bout[o];
#pragma unroll
  for (int r = 0; r < 8; ++r) {
    int t = tt * 16 + r + half * 8;
    size_t idx = ((size_t)(b * CCH + o)) * NTOK + t;
    out[idx] = acc[r] + bias + x[idx];
  }
}

// ---------------- Launch ---------------------------------------------------

extern "C" void kernel_launch(void* const* d_in, const int* in_sizes, int n_in,
                              void* d_out, int out_size, void* d_ws, size_t ws_size,
                              hipStream_t stream) {
  const float* x     = (const float*)d_in[0];
  const float* wqkv  = (const float*)d_in[1];
  const float* bqkv  = (const float*)d_in[2];
  const float* wout  = (const float*)d_in[3];
  const float* bout  = (const float*)d_in[4];
  const float* gamma = (const float*)d_in[5];
  const float* beta  = (const float*)d_in[6];
  float* out = (float*)d_out;

  char* ws = (char*)d_ws;
  size_t off = 0;
  float* scale = (float*)(ws + off); off += 1024;
  float* shift = (float*)(ws + off); off += 1024;
  _Float16* wqkv_h = (_Float16*)(ws + off); off += (size_t)C3 * CCH * 2;
  _Float16* wout_h = (_Float16*)(ws + off); off += (size_t)CCH * CCH * 2;
  _Float16* xn = (_Float16*)(ws + off); off += (size_t)BATCH * NTOK * CCH * 2;
  _Float16* qh = (_Float16*)(ws + off); off += (size_t)BATCH * NTOK * CCH * 2;
  _Float16* kh = (_Float16*)(ws + off); off += (size_t)BATCH * NTOK * CCH * 2;
  _Float16* vt = (_Float16*)(ws + off); off += (size_t)BATCH * NTOK * CCH * 2;
  _Float16* oh = (_Float16*)(ws + off); off += (size_t)BATCH * NTOK * CCH * 2;

  bn_stats_kernel<<<CCH, 256, 0, stream>>>(x, gamma, beta, scale, shift);
  cvt_w_kernel<<<(C3 * CCH) / 256, 256, 0, stream>>>(wqkv, wout, wqkv_h, wout_h);
  bn_transpose_kernel<<<dim3(NTOK / 32, CCH / 32, BATCH), dim3(32, 8), 0, stream>>>(
      x, scale, shift, xn);
  qkv_gemm_kernel<<<(BATCH * 48 * 256) / 8, 256, 0, stream>>>(
      xn, wqkv_h, bqkv, qh, kh, vt);
  flash_attn_kernel<<<(BATCH * 256) / 8, 256, 73728, stream>>>(qh, kh, vt, oh);
  out_proj_kernel<<<(BATCH * 16 * 256) / 8, 256, 0, stream>>>(
      oh, wout_h, bout, x, out);
}
